// groupedBatchedExperts_9457517986491
// MI455X (gfx1250) — matrix operand order repaired and checked by reading, not compiled, not run on hardware
//
#include <hip/hip_runtime.h>
#include <hip/hip_bf16.h>

// ---------------------------------------------------------------------------
// MoE grouped-expert MLP for gfx1250 (MI455X), wave32 + WMMA bf16.
//   T=2048 tokens, K=2 top-k, E=8 experts, H=1024 hidden, F=4096 ffn
// Pipeline:
//   0) zero out + counters
//   1) scatter tokens into per-expert compacted lists
//   2) pack w1/w2 -> bf16 in per-lane WMMA B-fragment layout (one pass)
//   3) fused MLP over gathered 32-token tiles. GEMM2 keeps 8 B-fragments
//      (16 global_load_b128) in flight per k-step; every B-fragment feeds
//      TWO v_wmma_f32_16x16x32_bf16 (two M-subtiles); atomic scatter-combine.
// ---------------------------------------------------------------------------

#define TT   2048
#define KTOP 2
#define NE   8
#define HD   1024
#define FD   4096
#define CAP  (TT * KTOP)   // max gathered entries per expert
#define MT   32            // gathered tokens per block (2 x 16-row subtiles)

// packed-weight geometry: [e][ntile][kchunk][lane(32)][v(8)] dwords
#define NT1  (FD / 16)     // 256 col tiles in w1
#define KC1  (HD / 32)     // 32  k-chunks in w1
#define NT2  (HD / 16)     // 64  col tiles in w2
#define KC2  (FD / 32)     // 128 f-chunks in w2

typedef __attribute__((ext_vector_type(16))) __bf16 v16bf;
typedef __attribute__((ext_vector_type(8)))  float  v8f;

union ABFrag {
    v16bf        v;
    unsigned int u[8];
    uint4        q[2];
};

// f32 pair -> packed bf16 dword (HW cvt_pk when available, RNE fallback)
__device__ __forceinline__ unsigned int cvtpk_bf16(float a, float b) {
#if __has_builtin(__builtin_amdgcn_cvt_pk_bf16_f32)
    auto r = __builtin_amdgcn_cvt_pk_bf16_f32(a, b);
    unsigned int u;
    __builtin_memcpy(&u, &r, 4);
    return u;
#else
    unsigned int ua = __float_as_uint(a), ub = __float_as_uint(b);
    unsigned int ra = (ua + 0x7FFFu + ((ua >> 16) & 1u)) >> 16;
    unsigned int rb = (ub + 0x7FFFu + ((ub >> 16) & 1u)) >> 16;
    return (ra & 0xFFFFu) | (rb << 16);
#endif
}

__device__ __forceinline__ unsigned short f2bf(float f) {
    return (unsigned short)(cvtpk_bf16(f, 0.0f) & 0xFFFFu);
}

// A-operand K offset for VGPR v, lane-half hi (ISA 7.12.2, 16-bit A 16x32)
__device__ __forceinline__ int kAoff(int v, int hi) {
    return ((v & 3) << 1) + ((v >> 2) << 4) + (hi << 3);
}

// ---------------------------------------------------------------------------
// Kernel 0: zero output accumulator + per-expert counters
// ---------------------------------------------------------------------------
__global__ void moe_zero_kernel(float* __restrict__ out, int n, int* __restrict__ counts) {
    int i = blockIdx.x * blockDim.x + threadIdx.x;
    if (i < n) out[i] = 0.0f;
    if (i < NE) counts[i] = 0;
}

// ---------------------------------------------------------------------------
// Kernel 1: scatter (token, k) entries into per-expert compacted lists
// ---------------------------------------------------------------------------
__global__ void moe_scatter_kernel(const int* __restrict__ top_experts,
                                   const float* __restrict__ expert_weights,
                                   int* __restrict__ counts,
                                   int* __restrict__ rows,
                                   float* __restrict__ wgts) {
    int i = blockIdx.x * blockDim.x + threadIdx.x;
    if (i >= TT * KTOP) return;
    int e = top_experts[i] & (NE - 1);
    int pos = atomicAdd(&counts[e], 1);
    rows[e * CAP + pos] = i / KTOP;
    wgts[e * CAP + pos] = expert_weights[i];
}

// ---------------------------------------------------------------------------
// Kernel 2: pack f32 weights [E][Kdim][Ndim] into bf16 B-fragment layout:
//   dst dword i = [e][tile][kc][lane][v],
//   pair ( src[e][kc*32 + 2*(v&3) + 16*(v>>2) + 8*hi][tile*16 + n], src[..+1][..] ),
//   hi = lane>>4, n = lane&15.  Each 1KB block is one 32x16 K-tile.
// ---------------------------------------------------------------------------
__global__ __launch_bounds__(256)
void moe_pack_kernel(const float* __restrict__ src, unsigned int* __restrict__ dst,
                     int Kdim, int Ndim) {
    int ntiles  = Ndim >> 4;
    int kchunks = Kdim >> 5;
    size_t dwPerTile = (size_t)kchunks << 8;
    size_t dwPerExp  = (size_t)ntiles * dwPerTile;
    size_t total     = (size_t)NE * dwPerExp;

    size_t i = (size_t)blockIdx.x * blockDim.x + threadIdx.x;
    if (i >= total) return;

    int e      = (int)(i / dwPerExp);
    size_t r   = i % dwPerExp;
    int tile   = (int)(r / dwPerTile);
    int r2     = (int)(r % dwPerTile);
    int kc     = r2 >> 8;
    int lane   = (r2 >> 3) & 31;
    int v      = r2 & 7;
    int hi     = lane >> 4;
    int n      = lane & 15;
    int k      = (kc << 5) + ((v & 3) << 1) + ((v >> 2) << 4) + (hi << 3);
    int col    = (tile << 4) + n;

    size_t s = ((size_t)e * Kdim + k) * Ndim + col;
    dst[i] = cvtpk_bf16(src[s], src[s + Ndim]);
}

// ---------------------------------------------------------------------------
// Kernel 3: fused expert MLP over gathered 32-token tiles.
//   block = 256 threads = 8 waves (wave32). grid = (CAP/MT, NE).
//   launch_bounds(256, 1): allow full VGPR budget so the scheduler can keep
//   many b128 loads in flight instead of recycling one B register set.
// ---------------------------------------------------------------------------
__global__ __launch_bounds__(256, 1)
void moe_mlp_kernel(const float* __restrict__ x,
                    const unsigned int* __restrict__ w1p,
                    const unsigned int* __restrict__ w2p,
                    const int* __restrict__ counts,
                    const int* __restrict__ rows,
                    const float* __restrict__ wgts,
                    float* __restrict__ out) {
    const int e    = blockIdx.y;
    const int n    = counts[e];
    const int base = blockIdx.x * MT;
    if (base >= n) return;   // tile beyond this expert's token count

    // x tile pre-swizzled into A-fragment layout, 2 row-subtiles:
    //   xa[t][kc(32)][lane(32)][v(8)] dwords  (2 x 32 KB)
    __shared__ __align__(16) unsigned int   xa[2 * KC1 * 256];   // 64 KB
    __shared__ __align__(16) unsigned short hs[MT * 128];        //  8 KB, plain [m][f]
    __shared__ int   rowIdx[MT];
    __shared__ float rowWs[MT];

    const int tid = threadIdx.x;
    if (tid < MT) {
        int idx = base + tid;
        bool ok = idx < n;
        rowIdx[tid] = ok ? rows[e * CAP + idx] : 0;
        rowWs[tid]  = ok ? wgts[e * CAP + idx] : 0.0f;   // padded rows contribute 0
    }
    __syncthreads();

    // Stage gathered x rows -> bf16 A-fragment layout (float2 + cvt_pk + b32 store)
    for (int i = tid; i < 2 * KC1 * 256; i += 256) {
        int t = i >> 13;                 // row-subtile 0/1
        int j = i & 8191;
        int kc = j >> 8, lane = (j >> 3) & 31, v = j & 7;
        int m  = (lane & 15) + (t << 4);
        int k  = (kc << 5) + kAoff(v, lane >> 4);
        const float2 p = *(const float2*)&x[(size_t)rowIdx[m] * HD + k];
        xa[i] = cvtpk_bf16(p.x, p.y);
    }
    __syncthreads();

    const int lane = tid & 31;
    const int wv   = tid >> 5;     // wave id 0..7
    const int mn   = lane & 15;
    const int hi   = lane >> 4;

    const unsigned int* __restrict__ w1e = w1p + ((size_t)e * NT1 * KC1 << 8);
    const unsigned int* __restrict__ w2e = w2p + ((size_t)e * NT2 * KC2 << 8);

    int kA[8];
#pragma unroll
    for (int v = 0; v < 8; ++v) kA[v] = kAoff(v, hi);

    // Output accumulators: wave owns 128 out columns x 32 rows = 2x8 16x16 tiles.
    v8f acc2[2][8];
#pragma unroll
    for (int t = 0; t < 2; ++t)
#pragma unroll
        for (int ct = 0; ct < 8; ++ct) acc2[t][ct] = v8f{0, 0, 0, 0, 0, 0, 0, 0};

    for (int f0 = 0; f0 < FD; f0 += 128) {
        // ---- GEMM1: h[32x16] = x_tile[32xH] @ w1[:, wave's 16 cols] ----
        const int ftile = (f0 >> 4) + wv;
        const unsigned int* __restrict__ b1base =
            w1e + (((size_t)ftile * KC1) << 8) + (lane << 3);
        if (f0 + 128 < FD)
            __builtin_prefetch(w1e + ((((size_t)ftile + 8) * KC1) << 8), 0, 1);

        v8f hacc0 = v8f{0, 0, 0, 0, 0, 0, 0, 0};
        v8f hacc1 = v8f{0, 0, 0, 0, 0, 0, 0, 0};
        for (int kcg = 0; kcg < KC1; kcg += 4) {
            // load 4 B-fragments first: one clause window of 8 b128 loads
            ABFrag b[4];
#pragma unroll
            for (int j = 0; j < 4; ++j) {
                const uint4* bp = (const uint4*)(b1base + ((size_t)(kcg + j) << 8));
                b[j].q[0] = bp[0];
                b[j].q[1] = bp[1];
            }
#pragma unroll
            for (int j = 0; j < 4; ++j) {
                ABFrag a0, a1;
                const uint4* ap0 = (const uint4*)&xa[((kcg + j) << 8) + (lane << 3)];
                a0.q[0] = ap0[0];                    // ds_read_b128 x2
                a0.q[1] = ap0[1];
                const uint4* ap1 = (const uint4*)&xa[8192 + ((kcg + j) << 8) + (lane << 3)];
                a1.q[0] = ap1[0];
                a1.q[1] = ap1[1];
                hacc0 = __builtin_amdgcn_wmma_f32_16x16x32_bf16(
                    false, a0.v, false, b[j].v, (short)0, hacc0, false, false);
                hacc1 = __builtin_amdgcn_wmma_f32_16x16x32_bf16(
                    false, a1.v, false, b[j].v, (short)0, hacc1, false, false);
            }
        }

        // ---- exact gelu, fold in combine weight, stash h as bf16 in LDS ----
#pragma unroll
        for (int v = 0; v < 8; ++v) {
            int r0 = v + (hi << 3);                  // C/D layout: M = v + 8*hi
            float t0 = hacc0[v];
            float g0 = 0.5f * t0 * (1.0f + erff(t0 * 0.70710678118654752f));
            hs[r0 * 128 + (wv << 4) + mn] = f2bf(g0 * rowWs[r0]);
            int r1 = r0 + 16;
            float t1 = hacc1[v];
            float g1 = 0.5f * t1 * (1.0f + erff(t1 * 0.70710678118654752f));
            hs[r1 * 128 + (wv << 4) + mn] = f2bf(g1 * rowWs[r1]);
        }
        __syncthreads();

        // ---- GEMM2: out[32x1024] += h[32x128] @ w2[f0:f0+128, :] ----
#pragma unroll
        for (int kkc = 0; kkc < 4; ++kkc) {
            ABFrag a20, a21;
            const int kk = kkc << 5;
#pragma unroll
            for (int v = 0; v < 8; ++v) {
                a20.u[v] = *(const unsigned int*)&hs[(mn << 7) + kk + kA[v]];
                a21.u[v] = *(const unsigned int*)&hs[((mn + 16) << 7) + kk + kA[v]];
            }

            const int kc2 = (f0 >> 5) + kkc;
            // load ALL 8 B-fragments of this k-step first (16 b128 in flight)
            ABFrag b2[8];
#pragma unroll
            for (int j = 0; j < 8; ++j) {
                const int ntile = (wv << 3) + j;
                const uint4* bp = (const uint4*)
                    (w2e + ((((size_t)ntile * KC2) + kc2) << 8) + (lane << 3));
                b2[j].q[0] = bp[0];
                b2[j].q[1] = bp[1];
            }
#pragma unroll
            for (int j = 0; j < 8; ++j) {
                acc2[0][j] = __builtin_amdgcn_wmma_f32_16x16x32_bf16(
                    false, a20.v, false, b2[j].v, (short)0, acc2[0][j], false, false);
                acc2[1][j] = __builtin_amdgcn_wmma_f32_16x16x32_bf16(
                    false, a21.v, false, b2[j].v, (short)0, acc2[1][j], false, false);
            }
        }
        __syncthreads();  // hs rewritten next f-chunk
    }

    // ---- scatter-add this tile's 32x1024 output (combine across experts) ----
#pragma unroll
    for (int t = 0; t < 2; ++t) {
#pragma unroll
        for (int ct = 0; ct < 8; ++ct) {
#pragma unroll
            for (int v = 0; v < 8; ++v) {
                int r   = v + (hi << 3) + (t << 4);
                int col = (wv << 7) + (ct << 4) + mn;
                atomicAdd(&out[(size_t)rowIdx[r] * HD + col], acc2[t][ct][v]);
            }
        }
    }
}

// ---------------------------------------------------------------------------
// Host launcher
// ---------------------------------------------------------------------------
extern "C" void kernel_launch(void* const* d_in, const int* in_sizes, int n_in,
                              void* d_out, int out_size, void* d_ws, size_t ws_size,
                              hipStream_t stream) {
    const float* x   = (const float*)d_in[0];
    // d_in[1] = scores (unused by reference math)
    const float* ew  = (const float*)d_in[2];
    const int*   top = (const int*)d_in[3];
    const float* w1  = (const float*)d_in[4];
    const float* w2  = (const float*)d_in[5];
    float* out = (float*)d_out;

    // workspace layout
    char* ws = (char*)d_ws;
    int*   counts = (int*)ws;                                     // 8 ints
    int*   rows   = (int*)(ws + 256);                             // 128 KB
    float* wgts   = (float*)(ws + 256 + NE * CAP * sizeof(int));  // 128 KB
    unsigned int* w1p = (unsigned int*)(ws + (1 << 20));                      // 64 MB
    unsigned int* w2p = (unsigned int*)(ws + (1 << 20) + (size_t)NE * HD * FD * 2); // 64 MB

    // 0) zero output + counters
    {
        int n = TT * HD;
        moe_zero_kernel<<<(n + 255) / 256, 256, 0, stream>>>(out, n, counts);
    }
    // 1) gather tokens per expert
    {
        moe_scatter_kernel<<<(TT * KTOP + 255) / 256, 256, 0, stream>>>(
            top, ew, counts, rows, wgts);
    }
    // 2) pack weights into bf16 WMMA fragment layout
    {
        size_t dw = (size_t)NE * HD * FD / 2;   // dwords per weight tensor
        int blocks = (int)((dw + 255) / 256);
        moe_pack_kernel<<<blocks, 256, 0, stream>>>(w1, w1p, HD, FD);
        moe_pack_kernel<<<blocks, 256, 0, stream>>>(w2, w2p, FD, HD);
    }
    // 3) fused expert MLP over gathered tiles
    {
        dim3 grid(CAP / MT, NE, 1);
        moe_mlp_kernel<<<grid, 256, 0, stream>>>(x, w1p, w2p, counts, rows, wgts, out);
    }
}
